// StableMoEGate_43928925503870
// MI455X (gfx1250) — matrix operand order repaired and checked by reading, compiled, not run on hardware
//
#include <hip/hip_runtime.h>
#include <stdint.h>

typedef __attribute__((ext_vector_type(16))) __bf16       v16bf;
typedef __attribute__((ext_vector_type(8)))  float        v8f;
typedef __attribute__((ext_vector_type(4)))  unsigned int v4u;
typedef __attribute__((ext_vector_type(8)))  unsigned int v8u;

// Problem constants (StableMoEGate: x [B=4,S=4096,H=4096], W [E=64,H=4096])
constexpr int kH  = 4096;
constexpr int kE  = 64;
constexpr int kBK = 64;                        // K-chunk per TDM load
constexpr int kKIters = kH / kBK;              // 64
constexpr int kF32Stride = 68;                 // dwords: 64 + 4 TDM pad
constexpr int kF32Tile   = 64 * kF32Stride;    // 4352 dw per staged fp32 tile
constexpr int kBfStride  = 36;                 // dwords: 64 bf16 = 32dw + 4 pad
constexpr int kBfTile    = 64 * kBfStride;     // 2304 dw per bf16 tile
// LDS map (dword offsets): fp32 x/W double-buffered TDM targets, bf16 single set
constexpr int XF0 = 0;
constexpr int XF1 = kF32Tile;
constexpr int WF0 = 2 * kF32Tile;
constexpr int WF1 = 3 * kF32Tile;
constexpr int XH  = 4 * kF32Tile;
constexpr int XL  = XH + kBfTile;
constexpr int WH  = XL + kBfTile;
constexpr int WL  = WH + kBfTile;
constexpr int kSmemDW = WL + kBfTile;          // 26624 dw = 104 KB

union BFU2 { unsigned d[8]; v16bf v; };

// Issue one 2D TDM load: rows x 64 fp32 tile, memory row stride kH elements,
// LDS rows padded +4 dwords per 64 dwords (row stride 68 dw).
__device__ __forceinline__ void tdm_load_2d(unsigned lds_byte_addr, const float *gptr,
                                            unsigned rows) {
  unsigned long long ga = (unsigned long long)(uintptr_t)gptr;
  v4u g0;
  g0[0] = 1u;                                                  // count=1, user mode
  g0[1] = lds_byte_addr;                                       // lds_addr
  g0[2] = (unsigned)ga;                                        // global_addr[31:0]
  g0[3] = (unsigned)((ga >> 32) & 0x01ffffffu) | 0x80000000u;  // addr[56:32] | type=2
  v8u g1;
  g1[0] = (2u << 16)        // data_size = 4B
        | (1u << 20)        // pad_enable
        | (5u << 22)        // pad_interval code 5 -> every 64 dwords
        | (3u << 25);       // pad_amount  code 3 -> 4 dwords
  g1[1] = ((unsigned)kH & 0xffffu) << 16;                                // dim0 lo16
  g1[2] = ((rows & 0xffffu) << 16) | (((unsigned)kH >> 16) & 0xffffu);   // dim1 lo | dim0 hi
  g1[3] = ((unsigned)kBK << 16) | ((rows >> 16) & 0xffffu);              // tile0 | dim1 hi
  g1[4] = rows;                                                          // tile1 (tile2=0)
  g1[5] = (unsigned)kH;                                                  // stride0 lo32
  g1[6] = 0u;
  g1[7] = 0u;
  asm volatile("tensor_load_to_lds %0, %1" :: "s"(g0), "s"(g1) : "memory");
}

// Cooperative fp32 -> (bf16 hi, bf16 lo) tile conversion: 64 threads, 64x64 tile,
// one full row per thread. hi = truncate-top16, lo = bf16(f - hi). Packing via
// v_perm_b32: ~3 VALU per element, no 16-bit register shuffles.
__device__ __forceinline__ void convert_tile(const float* __restrict__ f32t,
                                             unsigned* __restrict__ hi_t,
                                             unsigned* __restrict__ lo_t, int tid) {
  const float* src = f32t + tid * kF32Stride;   // 16B aligned (68 % 4 == 0)
  unsigned* dh = hi_t + tid * kBfStride;
  unsigned* dl = lo_t + tid * kBfStride;
#pragma unroll
  for (int j = 0; j < 16; ++j) {                // 4 elements per step
    const float4 f = *(const float4*)(src + 4 * j);
    const unsigned u0 = __float_as_uint(f.x), u1 = __float_as_uint(f.y);
    const unsigned u2 = __float_as_uint(f.z), u3 = __float_as_uint(f.w);
    const float r0 = f.x - __uint_as_float(u0 & 0xffff0000u);
    const float r1 = f.y - __uint_as_float(u1 & 0xffff0000u);
    const float r2 = f.z - __uint_as_float(u2 & 0xffff0000u);
    const float r3 = f.w - __uint_as_float(u3 & 0xffff0000u);
    uint2 hp, lp;
    hp.x = __builtin_amdgcn_perm(u1, u0, 0x07060302u);  // [bf16(f1)|bf16(f0)]
    hp.y = __builtin_amdgcn_perm(u3, u2, 0x07060302u);
    lp.x = __builtin_amdgcn_perm(__float_as_uint(r1), __float_as_uint(r0), 0x07060302u);
    lp.y = __builtin_amdgcn_perm(__float_as_uint(r3), __float_as_uint(r2), 0x07060302u);
    *(uint2*)(dh + 2 * j) = hp;
    *(uint2*)(dl + 2 * j) = lp;
  }
}

__global__ __launch_bounds__(64, 1)
void moe_gate_kernel(const float* __restrict__ x, const float* __restrict__ W,
                     float* __restrict__ out, int T) {
  __shared__ float smemf[kSmemDW];
  const int tid  = threadIdx.x;
  const int wave = tid >> 5;       // 2 waves per block
  const int lane = tid & 31;
  const int hh   = lane >> 4;      // WMMA lane half
  const int nlo  = lane & 15;
  const int blk  = blockIdx.x;
  const unsigned lds_base = (unsigned)(uintptr_t)(void*)&smemf[0];
  const bool wave0 = (wave == 0);

  // Each wave owns 2 M-tiles (32 tokens); block = 64 tokens.
  v8f acc[2][4];
#pragma unroll
  for (int m = 0; m < 2; ++m)
#pragma unroll
    for (int t = 0; t < 4; ++t)
#pragma unroll
      for (int i = 0; i < 8; ++i) acc[m][t][i] = 0.0f;

  const float* xblk = x + (size_t)blk * 64 * kH;

  if (wave0) {                                   // prologue: stage chunk 0
    tdm_load_2d(lds_base + XF0 * 4, xblk, 64u);
    tdm_load_2d(lds_base + WF0 * 4, W, 64u);
  }

  unsigned* const xh = (unsigned*)&smemf[XH];
  unsigned* const xl = (unsigned*)&smemf[XL];
  unsigned* const wh = (unsigned*)&smemf[WH];
  unsigned* const wl = (unsigned*)&smemf[WL];

  for (int it = 0; it < kKIters; ++it) {
    if (wave0) {
      if (it + 1 < kKIters) {
        const int k0n = (it + 1) * kBK;
        const int bn  = (it + 1) & 1;
        tdm_load_2d(lds_base + (bn ? XF1 : XF0) * 4, xblk + k0n, 64u);
        tdm_load_2d(lds_base + (bn ? WF1 : WF0) * 4, W + k0n, 64u);
        __builtin_amdgcn_s_wait_tensorcnt(2);    // this iter's 2 tiles landed
      } else {
        __builtin_amdgcn_s_wait_tensorcnt(0);
      }
    }
    __syncthreads();  // fp32 tiles visible; previous compute done (bf16 bufs free)

    const float* xf = &smemf[(it & 1) ? XF1 : XF0];
    const float* wf = &smemf[(it & 1) ? WF1 : WF0];
    convert_tile(xf, xh, xl, tid);
    convert_tile(wf, wh, wl, tid);
    __syncthreads();  // bf16 tiles ready

#pragma unroll
    for (int s = 0; s < 2; ++s) {                // two K=32 substeps
      const int kd = s * 16;                     // dword base (= k/2)
      BFU2 ah[2], al[2];                         // A frags for 2 M-tiles
#pragma unroll
      for (int m = 0; m < 2; ++m) {
        const unsigned* arow_h = xh + (wave * 32 + m * 16 + nlo) * kBfStride;
        const unsigned* arow_l = xl + (wave * 32 + m * 16 + nlo) * kBfStride;
        *(uint4*)&ah[m].d[0] = *(const uint4*)(arow_h + kd + 4 * hh);
        *(uint4*)&ah[m].d[4] = *(const uint4*)(arow_h + kd + 8 + 4 * hh);
        *(uint4*)&al[m].d[0] = *(const uint4*)(arow_l + kd + 4 * hh);
        *(uint4*)&al[m].d[4] = *(const uint4*)(arow_l + kd + 8 + 4 * hh);
      }
#pragma unroll
      for (int t = 0; t < 4; ++t) {              // 4 expert tiles of 16
        const unsigned* brow_h = wh + (nlo + 16 * t) * kBfStride + kd + 4 * hh;
        const unsigned* brow_l = wl + (nlo + 16 * t) * kBfStride + kd + 4 * hh;
        BFU2 bh, bl;                             // B frag: k 8h..8h+7, 16+8h..16+8h+7
        *(uint4*)&bh.d[0] = *(const uint4*)(brow_h);
        *(uint4*)&bh.d[4] = *(const uint4*)(brow_h + 8);
        *(uint4*)&bl.d[0] = *(const uint4*)(brow_l);
        *(uint4*)&bl.d[4] = *(const uint4*)(brow_l + 8);
#pragma unroll
        for (int m = 0; m < 2; ++m) {
          // 3-product bf16 split ~= fp32 GEMM: hi*hi + hi*lo + lo*hi
          acc[m][t] = __builtin_amdgcn_wmma_f32_16x16x32_bf16(false, ah[m].v, false, bh.v,
                                                              (short)0, acc[m][t], false, false);
          acc[m][t] = __builtin_amdgcn_wmma_f32_16x16x32_bf16(false, ah[m].v, false, bl.v,
                                                              (short)0, acc[m][t], false, false);
          acc[m][t] = __builtin_amdgcn_wmma_f32_16x16x32_bf16(false, al[m].v, false, bh.v,
                                                              (short)0, acc[m][t], false, false);
        }
      }
    }
  }

  // Scatter logits [64 tokens x 64 experts] into smemf[0..] (stride 68 dw).
#pragma unroll
  for (int m = 0; m < 2; ++m)
#pragma unroll
    for (int t = 0; t < 4; ++t)
#pragma unroll
      for (int r = 0; r < 8; ++r) {
        const int tok = wave * 32 + m * 16 + (hh << 3) + r;  // token within block
        const int e   = nlo + 16 * t;                        // expert
        smemf[tok * kF32Stride + e] = acc[m][t][r];
      }
  __syncthreads();

  // One thread per token: softmax over 64, top-2 (lowest-index tie-break),
  // then renormalize the two probabilities with a 2-way softmax.
  {
    const float* row = &smemf[tid * kF32Stride];
    float mx = row[0];
#pragma unroll 8
    for (int e = 1; e < kE; ++e) mx = fmaxf(mx, row[e]);
    float sum = 0.0f, b1 = -3.402823466e38f, b2 = -3.402823466e38f;
    int i1 = 0, i2 = 0;
#pragma unroll 4
    for (int e = 0; e < kE; ++e) {
      const float v = row[e];
      sum += __expf(v - mx);
      if (v > b1)      { b2 = b1; i2 = i1; b1 = v; i1 = e; }
      else if (v > b2) { b2 = v; i2 = e; }
    }
    const float inv = 1.0f / sum;
    const float s1 = __expf(b1 - mx) * inv;
    const float s2 = __expf(b2 - mx) * inv;
    const float e2 = __expf(s2 - s1);
    const float o1 = 1.0f / (1.0f + e2);
    const float o2 = e2 * o1;
    const int tok = blk * 64 + tid;
    out[2 * tok]     = o1;
    out[2 * tok + 1] = o2;
    int* oidx = (int*)(out + 2 * (size_t)T);     // int32 bit patterns
    oidx[2 * tok]     = i1;
    oidx[2 * tok + 1] = i2;
  }
  if (blk == 0 && tid == 0) out[4 * (size_t)T] = 0.0f;  // aux_loss = 0.0
}

extern "C" void kernel_launch(void* const* d_in, const int* in_sizes, int n_in,
                              void* d_out, int out_size, void* d_ws, size_t ws_size,
                              hipStream_t stream) {
  (void)n_in; (void)out_size; (void)d_ws; (void)ws_size;
  const float* x = (const float*)d_in[0];
  const float* W = (const float*)d_in[1];
  float* out = (float*)d_out;
  const int T = in_sizes[0] / kH;   // 16384 tokens
  const int blocks = T / 64;        // 256 blocks of 2 wave32 waves, 64 tokens each
  moe_gate_kernel<<<blocks, 64, 0, stream>>>(x, W, out, T);
}
